// GCN_42941083025543
// MI455X (gfx1250) — matrix operand order repaired and checked by reading, not compiled, not run on hardware
//
#include <hip/hip_runtime.h>
#include <hip/hip_bf16.h>

#define IN_CH   512
#define OUT_CH  256

typedef __attribute__((ext_vector_type(16))) __bf16 v16bf;
typedef __attribute__((ext_vector_type(8)))  __bf16 v8bf;
typedef __attribute__((ext_vector_type(8)))  float  v8f;

// ---------------------------------------------------------------------------
// Kernel 1: split W[512][256] (f32, row-major k,n) into bf16 hi/lo, stored
// TRANSPOSED as [n][k] so the WMMA B-fragment (K kbB..+7, kbB+16..+23 per lane) is
// two contiguous 16B loads (global_load_b128).
// ---------------------------------------------------------------------------
__global__ void gcn_split_w(const float* __restrict__ W,
                            __bf16* __restrict__ whi_t,
                            __bf16* __restrict__ wlo_t) {
    int i = blockIdx.x * 256 + threadIdx.x;        // 512*256 elements
    int k = i >> 8;          // 0..511
    int n = i & 255;         // 0..255
    float w  = W[(size_t)k * OUT_CH + n];
    __bf16 hi = (__bf16)w;                         // RNE
    __bf16 lo = (__bf16)(w - (float)hi);
    whi_t[(size_t)n * IN_CH + k] = hi;
    wlo_t[(size_t)n * IN_CH + k] = lo;
}

// ---------------------------------------------------------------------------
// Kernel 2: zero output (harness poisons d_out with 0xAA).
// ---------------------------------------------------------------------------
__global__ void gcn_zero(float4* __restrict__ out, int n4) {
    int i = blockIdx.x * 256 + threadIdx.x;
    if (i < n4) out[i] = make_float4(0.f, 0.f, 0.f, 0.f);
}

// ---------------------------------------------------------------------------
// Kernel 3: z = x @ W via split-bf16 WMMA (v_wmma_f32_16x16x32_bf16).
// Block: 256 threads = 8 waves; block tile = 64 rows x 256 cols.
// Wave w: rows (w>>2)*32 .. +31 (two 16-row tiles), cols (w&3)*64 .. +63
// (four 16-col tiles) -> eight 16x16 C accumulators, 24 WMMAs per k-step.
// Each B hi/lo fragment pair feeds 6 WMMAs (2 independent row chains).
// x tile staged f32->bf16(hi,lo) via double-buffered LDS, 1 barrier/k-step.
// ---------------------------------------------------------------------------
__global__ void __launch_bounds__(256)
gcn_gemm_wmma(const float* __restrict__ x,
              const __bf16* __restrict__ whi_t,
              const __bf16* __restrict__ wlo_t,
              float* __restrict__ z, int nodes) {
    // [buffer][hi=0/lo=1][64 rows * 32 k] bf16  -> 16 KB total
    __shared__ __bf16 sA[2][2][64 * 32];

    const int lane = threadIdx.x & 31;
    const int wave = threadIdx.x >> 5;
    const int m0   = blockIdx.x * 64;

    const int rwave = (wave >> 2) * 32;         // wave's 32-row slice in tile
    const int colq  = wave & 3;                 // wave's 64-col slice

    // Per-lane fragment geometry (ISA 7.12.2, 16-bit A 16x32 / B 32x16):
    const int kbA  = (lane & 16) ? 8 : 0;       // A: K {kbA..+7, kbA+16..+23}
    const int ncol = lane & 15;                 // B col within 16-tile
    const int kbB  = (lane & 16) ? 8 : 0;       // B: K {kbB..+7, kbB+16..+23}

    // Cooperative stage geometry: 256 threads x 8 f32 = 64x32 tile
    const int sidx = threadIdx.x * 8;
    const int srow = sidx >> 5;                 // 0..63
    const int skk  = sidx & 31;                 // multiple of 8

    v8f acc[2][4] = {{{}, {}, {}, {}}, {{}, {}, {}, {}}};

    // stage one 64x32 f32 chunk of x into LDS buffer `buf` as bf16 hi/lo
    auto stage = [&](int k0, int buf) {
        int gr = m0 + srow;
        if (gr >= nodes) gr = nodes - 1;        // clamp (last block only)
        const float* src = x + (size_t)gr * IN_CH + k0 + skk;
        const float4 f0 = *(const float4*)(src);
        const float4 f1 = *(const float4*)(src + 4);
        __bf16 h0 = (__bf16)f0.x, h1 = (__bf16)f0.y;
        __bf16 h2 = (__bf16)f0.z, h3 = (__bf16)f0.w;
        __bf16 h4 = (__bf16)f1.x, h5 = (__bf16)f1.y;
        __bf16 h6 = (__bf16)f1.z, h7 = (__bf16)f1.w;
        v8bf hv = {h0, h1, h2, h3, h4, h5, h6, h7};
        v8bf lv = {(__bf16)(f0.x - (float)h0), (__bf16)(f0.y - (float)h1),
                   (__bf16)(f0.z - (float)h2), (__bf16)(f0.w - (float)h3),
                   (__bf16)(f1.x - (float)h4), (__bf16)(f1.y - (float)h5),
                   (__bf16)(f1.z - (float)h6), (__bf16)(f1.w - (float)h7)};
        *(v8bf*)&sA[buf][0][srow * 32 + skk] = hv;   // ds_store_b128
        *(v8bf*)&sA[buf][1][srow * 32 + skk] = lv;   // ds_store_b128
    };

    stage(0, 0);
    __syncthreads();

    int p = 0;
    for (int k0 = 0; k0 < IN_CH; k0 += 32) {
        // prefetch next K chunk into the other buffer (overlaps with WMMAs)
        if (k0 + 32 < IN_CH) stage(k0 + 32, p ^ 1);

        // ---- A fragments (2 row tiles x hi/lo) from LDS, ds_read_b128 x8 ----
        union frag { v16bf v; v8bf h[2]; };
        frag ahi[2], alo[2];
#pragma unroll
        for (int rt = 0; rt < 2; ++rt) {
            const int arow = rwave + rt * 16 + (lane & 15);
            const __bf16* ap = &sA[p][0][arow * 32 + kbA];
            ahi[rt].h[0] = *(const v8bf*)(ap);
            ahi[rt].h[1] = *(const v8bf*)(ap + 16);
            const __bf16* lp = &sA[p][1][arow * 32 + kbA];
            alo[rt].h[0] = *(const v8bf*)(lp);
            alo[rt].h[1] = *(const v8bf*)(lp + 16);
        }

        // ---- 4 col tiles: B frags from global (WGP$/L2-hot), 6 WMMAs each ----
#pragma unroll
        for (int t = 0; t < 4; ++t) {
            const int col = colq * 64 + t * 16 + ncol;
            frag bhi, blo;
            const __bf16* bp = whi_t + (size_t)col * IN_CH + k0 + kbB;
            bhi.h[0] = *(const v8bf*)(bp);
            bhi.h[1] = *(const v8bf*)(bp + 16);
            const __bf16* cp = wlo_t + (size_t)col * IN_CH + k0 + kbB;
            blo.h[0] = *(const v8bf*)(cp);
            blo.h[1] = *(const v8bf*)(cp + 16);

            // split-bf16 fp32 emulation: hi*hi + hi*lo + lo*hi
#pragma unroll
            for (int rt = 0; rt < 2; ++rt) {
                acc[rt][t] = __builtin_amdgcn_wmma_f32_16x16x32_bf16(
                    false, ahi[rt].v, false, bhi.v, (short)0, acc[rt][t], false, false);
                acc[rt][t] = __builtin_amdgcn_wmma_f32_16x16x32_bf16(
                    false, ahi[rt].v, false, blo.v, (short)0, acc[rt][t], false, false);
                acc[rt][t] = __builtin_amdgcn_wmma_f32_16x16x32_bf16(
                    false, alo[rt].v, false, bhi.v, (short)0, acc[rt][t], false, false);
            }
        }

        __syncthreads();     // staging of (k0+32) complete; reads of buf p done
        p ^= 1;
    }

    // ---- store C (ISA 7.12.2 32-bit C/D layout): VGPR v -> row v + (lane/16)*8
    const bool full = (m0 + 64 <= nodes);
#pragma unroll
    for (int rt = 0; rt < 2; ++rt) {
        const int rbase = m0 + rwave + rt * 16 + ((lane >> 4) << 3);
#pragma unroll
        for (int t = 0; t < 4; ++t) {
            const int ccol = colq * 64 + t * 16 + (lane & 15);
#pragma unroll
            for (int v = 0; v < 8; ++v) {
                if (full || (rbase + v) < nodes)
                    z[(size_t)(rbase + v) * OUT_CH + ccol] = acc[rt][t][v];
            }
        }
    }
}

// ---------------------------------------------------------------------------
// Kernel 4: out[row] += val * z[col]  (segment_sum via L2 f32 atomics).
// 64 threads per edge, float4 per thread. z and out are L2-resident (102MB
// each < 192MB L2), so gather + atomic RMW traffic stays on-die.
// ---------------------------------------------------------------------------
__global__ void gcn_scatter(const float* __restrict__ z,
                            const int* __restrict__ erow,
                            const int* __restrict__ ecol,
                            const float* __restrict__ eval,
                            float* __restrict__ out, int E) {
    int i = blockIdx.x * 256 + threadIdx.x;
    int e = i >> 6;
    if (e >= E) return;
    int q = (i & 63) << 2;                    // channel base 0..252
    int row = erow[e];
    int colv = ecol[e];
    float v = eval[e];
    const float4 zv = *(const float4*)(z + (size_t)colv * OUT_CH + q);
    float* o = out + (size_t)row * OUT_CH + q;
    atomicAdd(o + 0, v * zv.x);
    atomicAdd(o + 1, v * zv.y);
    atomicAdd(o + 2, v * zv.z);
    atomicAdd(o + 3, v * zv.w);
}

// ---------------------------------------------------------------------------
extern "C" void kernel_launch(void* const* d_in, const int* in_sizes, int n_in,
                              void* d_out, int out_size, void* d_ws, size_t ws_size,
                              hipStream_t stream) {
    const float* x    = (const float*)d_in[0];   // [N, 512]
    const float* W    = (const float*)d_in[1];   // [512, 256]
    const int*   erow = (const int*)d_in[2];     // [E]
    const int*   ecol = (const int*)d_in[3];     // [E]
    const float* eval = (const float*)d_in[4];   // [E]
    float* out = (float*)d_out;

    const int nodes = in_sizes[0] / IN_CH;       // 100000
    const int E     = in_sizes[2];               // 3200000

    // workspace: W_hi_t | W_lo_t | z
    char* ws = (char*)d_ws;
    __bf16* whi_t = (__bf16*)ws;
    __bf16* wlo_t = (__bf16*)(ws + (size_t)OUT_CH * IN_CH * sizeof(__bf16));
    float*  z     = (float*)(ws + 2 * (size_t)OUT_CH * IN_CH * sizeof(__bf16));

    // 1) split + transpose W
    gcn_split_w<<<(IN_CH * OUT_CH) / 256, 256, 0, stream>>>(W, whi_t, wlo_t);

    // 2) zero output
    {
        int n4 = (nodes * OUT_CH) / 4;
        gcn_zero<<<(n4 + 255) / 256, 256, 0, stream>>>((float4*)out, n4);
    }

    // 3) WMMA GEMM: z = x @ W   (64-row x 256-col block tiles, ceil-div grid)
    gcn_gemm_wmma<<<(nodes + 63) / 64, 256, 0, stream>>>(x, whi_t, wlo_t, z, nodes);

    // 4) edge scatter with f32 atomics
    {
        long long threads = (long long)E * 64;
        int blocks = (int)((threads + 255) / 256);
        gcn_scatter<<<blocks, 256, 0, stream>>>(z, erow, ecol, eval, out, E);
    }
}
